// VanillaRNN_90589450207611
// MI455X (gfx1250) — matrix operand order repaired and checked by reading, not compiled, not run on hardware
//
#include <hip/hip_runtime.h>
#include <hip/hip_bf16.h>

#define HDIM 1024
#define BDIM 256
#define TDIM 1024
#define CDIM 10

typedef __attribute__((ext_vector_type(16))) __bf16 v16bf;
typedef __attribute__((ext_vector_type(8)))  __bf16 v8bf;
typedef __attribute__((ext_vector_type(8)))  float  v8f;

// One-time prep: Wt[n][k] = (bf16)W[k][n]  (so the WMMA B-operand is a
// contiguous per-lane run), and copy W -> first tuple output (f32).
__global__ void prep_kernel(const float* __restrict__ W,
                            __bf16* __restrict__ Wt,
                            float* __restrict__ outW) {
    int tid = blockIdx.x * blockDim.x + threadIdx.x;   // [0, H*H)
    int n = tid >> 10;
    int k = tid & (HDIM - 1);
    Wt[tid]  = (__bf16)W[k * HDIM + n];
    outW[tid] = W[tid];
}

__global__ void zero_h_kernel(__bf16* __restrict__ h) {
    int tid = blockIdx.x * blockDim.x + threadIdx.x;
    h[tid] = (__bf16)0.0f;
}

// One timestep: h_out = tanh(x[:,t]*U^T + h_in @ W + bh)
// Grid: (H/128, B/16); block: 256 threads = 8 waves.
// Wave w: cw = w&3 selects a 32-col pair inside the 128-col block,
//         kh = w>>2 selects K half [kh*512, kh*512+512).
// K halves are combined via LDS; kh==0 waves finalize (bias+tanh+store).
__global__ __launch_bounds__(256)
void rnn_step_kernel(const __bf16* __restrict__ hin,   // [B][H] bf16
                     const __bf16* __restrict__ Wt,    // [H][H] bf16 transposed
                     const float*  __restrict__ x,     // [B][T]
                     const float*  __restrict__ U,     // [H]
                     const float*  __restrict__ bh,    // [H]
                     __bf16*       __restrict__ hout,  // [B][H] bf16
                     float*        __restrict__ hfinal,// [B][H] f32 (last step)
                     int t, int last) {
    __shared__ float red[4 * 2 * 8 * 32];  // 8 KB: partials from kh==1 waves

    const int lane = threadIdx.x & 31;
    const int wave = threadIdx.x >> 5;
    const int cw   = wave & 3;
    const int kh   = wave >> 2;
    const int rowbase = blockIdx.y * 16;
    const int colbase = blockIdx.x * 128 + cw * 32;
    const int m  = lane & 15;
    const int hi = lane >> 4;

    v8f acc0 = {};
    v8f acc1 = {};

    // A: row (rowbase+m), K half kh; lane holds 16 contiguous K at kb+hi*16
    //    K = {kb+hi*16 .. +7} and {kb+hi*16+8 .. +7}  (two 16B chunks)
    const __bf16* arow  = hin + (size_t)(rowbase + m) * HDIM + kh * 512;
    // B: column n = colbase(+16) + m; lane holds 16 contiguous K at kb+hi*16
    const __bf16* brow0 = Wt + (size_t)(colbase + m)      * HDIM + kh * 512 + hi * 16;
    const __bf16* brow1 = Wt + (size_t)(colbase + 16 + m) * HDIM + kh * 512 + hi * 16;

    #pragma unroll 4
    for (int kc = 0; kc < 512; kc += 32) {
        v8bf a0 = *(const v8bf*)(arow + kc + hi * 16);
        v8bf a1 = *(const v8bf*)(arow + kc + hi * 16 + 8);
        v16bf a = __builtin_shufflevector(a0, a1,
                  0,1,2,3,4,5,6,7,8,9,10,11,12,13,14,15);

        v8bf b00 = *(const v8bf*)(brow0 + kc);
        v8bf b01 = *(const v8bf*)(brow0 + kc + 8);
        v16bf b0 = __builtin_shufflevector(b00, b01,
                   0,1,2,3,4,5,6,7,8,9,10,11,12,13,14,15);

        v8bf b10 = *(const v8bf*)(brow1 + kc);
        v8bf b11 = *(const v8bf*)(brow1 + kc + 8);
        v16bf b1 = __builtin_shufflevector(b10, b11,
                   0,1,2,3,4,5,6,7,8,9,10,11,12,13,14,15);

        acc0 = __builtin_amdgcn_wmma_f32_16x16x32_bf16(
                   false, a, false, b0, (short)0, acc0, false, false);
        acc1 = __builtin_amdgcn_wmma_f32_16x16x32_bf16(
                   false, a, false, b1, (short)0, acc1, false, false);
    }

    if (kh == 1) {
        #pragma unroll
        for (int i = 0; i < 8; ++i) {
            red[((cw * 2 + 0) * 8 + i) * 32 + lane] = acc0[i];
            red[((cw * 2 + 1) * 8 + i) * 32 + lane] = acc1[i];
        }
    }
    __syncthreads();
    if (kh == 0) {
        #pragma unroll
        for (int i = 0; i < 8; ++i) {
            acc0[i] += red[((cw * 2 + 0) * 8 + i) * 32 + lane];
            acc1[i] += red[((cw * 2 + 1) * 8 + i) * 32 + lane];
        }
        const int n0 = colbase + m;
        const int n1 = colbase + 16 + m;
        const float u0 = U[n0],  u1 = U[n1];
        const float b0 = bh[n0], b1 = bh[n1];
        #pragma unroll
        for (int i = 0; i < 8; ++i) {
            const int row = rowbase + hi * 8 + i;   // C/D layout: M = i + 8*hi
            const float xv = x[(size_t)row * TDIM + t];
            const float h0 = tanhf(acc0[i] + xv * u0 + b0);
            const float h1 = tanhf(acc1[i] + xv * u1 + b1);
            hout[(size_t)row * HDIM + n0] = (__bf16)h0;
            hout[(size_t)row * HDIM + n1] = (__bf16)h1;
            if (last) {
                hfinal[(size_t)row * HDIM + n0] = h0;
                hfinal[(size_t)row * HDIM + n1] = h1;
            }
        }
    }
}

// output[b][c] = h_final[b] . V[c] + bp[c]   (tiny: 5 MFLOP, f32)
__global__ void classifier_kernel(const float* __restrict__ hf,
                                  const float* __restrict__ V,
                                  const float* __restrict__ bp,
                                  float* __restrict__ out) {
    int tid = blockIdx.x * blockDim.x + threadIdx.x;
    if (tid >= BDIM * CDIM) return;
    int b = tid / CDIM;
    int c = tid % CDIM;
    const float* hr = hf + (size_t)b * HDIM;
    const float* vr = V  + (size_t)c * HDIM;
    float acc = bp[c];
    for (int h = 0; h < HDIM; ++h) acc += hr[h] * vr[h];
    out[tid] = acc;
}

extern "C" void kernel_launch(void* const* d_in, const int* in_sizes, int n_in,
                              void* d_out, int out_size, void* d_ws, size_t ws_size,
                              hipStream_t stream) {
    const float* x  = (const float*)d_in[0];   // [B,T]
    const float* U  = (const float*)d_in[1];   // [H,1]
    const float* bh = (const float*)d_in[2];   // [H]
    const float* W  = (const float*)d_in[3];   // [H,H]
    const float* V  = (const float*)d_in[4];   // [C,H]
    const float* bp = (const float*)d_in[5];   // [C]

    // d_out = [ W (H*H) | h_final (B*H) | output (B*C) ], all f32
    float* outW  = (float*)d_out;
    float* outHf = outW + (size_t)HDIM * HDIM;
    float* outP  = outHf + (size_t)BDIM * HDIM;

    // Workspace: Wt bf16 (2 MB) + two bf16 h ping-pong buffers (0.5 MB each)
    char* ws = (char*)d_ws;
    __bf16* Wt = (__bf16*)ws;
    __bf16* hA = (__bf16*)(ws + (size_t)HDIM * HDIM * sizeof(__bf16));
    __bf16* hB = hA + (size_t)BDIM * HDIM;

    prep_kernel<<<(HDIM * HDIM) / 256, 256, 0, stream>>>(W, Wt, outW);
    zero_h_kernel<<<(BDIM * HDIM) / 256, 256, 0, stream>>>(hA);

    dim3 grid(HDIM / 128, BDIM / 16);
    for (int t = 0; t < TDIM; ++t) {
        const __bf16* hin = (t & 1) ? hB : hA;
        __bf16* hout      = (t & 1) ? hA : hB;
        rnn_step_kernel<<<grid, 256, 0, stream>>>(
            hin, Wt, x, U, bh, hout, outHf, t, (t == TDIM - 1) ? 1 : 0);
    }

    classifier_kernel<<<(BDIM * CDIM + 255) / 256, 256, 0, stream>>>(
        outHf, V, bp, outP);
}
